// CMDAutoEncoder_28020366639638
// MI455X (gfx1250) — matrix operand order repaired and checked by reading, not compiled, not run on hardware
//
#include <hip/hip_runtime.h>
#include <hip/hip_bf16.h>

typedef __attribute__((ext_vector_type(16))) _Float16 v16h;
typedef __attribute__((ext_vector_type(8)))  _Float16 v8h;
typedef __attribute__((ext_vector_type(8)))  float    v8f;

#define B_SZ      8
#define L_IN      16384
#define LC        2047            // (16384-16)/8 + 1
#define MROWS     (B_SZ * LC)     // 16376
#define MPAD      16384           // MROWS padded to 16-multiple (tile aligned)
#define NPROJ     1288            // D_IN_PROJ
#define NPROJ_PAD 1344            // padded to 64-multiple
#define DINNER    512
#define CONVDIM   768
#define DSTATE    128
#define NHEADS    8
#define HEADDIM   64

// ---------------------------------------------------------------------------
// K1: front conv1d (4->256, K=16, stride 8, VALID) + ReLU.
// Writes f16 GEMM-A operand, row-major (b*LC + l, 256).
// ---------------------------------------------------------------------------
__global__ void k_conv_front(const float* __restrict__ x,
                             const float* __restrict__ w,
                             const float* __restrict__ bconv,
                             _Float16* __restrict__ h16) {
    int idx = blockIdx.x * blockDim.x + threadIdx.x;
    if (idx >= MROWS * 256) return;
    int c  = idx & 255;
    int bl = idx >> 8;
    int l  = bl % LC;
    int b  = bl / LC;
    const float* xp = x + (size_t)b * 4 * L_IN + l * 8;
    const float* wp = w + (size_t)c * 64;
    float s = bconv[c];
#pragma unroll
    for (int ic = 0; ic < 4; ++ic)
#pragma unroll
        for (int k = 0; k < 16; ++k)
            s = fmaf(xp[ic * L_IN + k], wp[ic * 16 + k], s);
    h16[(size_t)bl * 256 + c] = (_Float16)fmaxf(s, 0.f);
}

__global__ void k_zero_f16(_Float16* __restrict__ p, int count) {
    int idx = blockIdx.x * blockDim.x + threadIdx.x;
    if (idx < count) p[idx] = (_Float16)0.f;
}

// ---------------------------------------------------------------------------
// Pre-swizzle W (N x K fp32 row-major) into f16 WMMA B-fragment order:
// fragment (ntile, ktile) is 512 contiguous f16; lane l owns [l*16, l*16+16).
// Per ISA 7.12.2: n = ntile*16 + (lane&15); k = ktile*32 + (j>>3)*16 + (lane>>4)*8 + (j&7).
// ---------------------------------------------------------------------------
__global__ void k_swizzleB(const float* __restrict__ W,
                           _Float16* __restrict__ Bsw,
                           int N, int K, int NPAD) {
    int idx = blockIdx.x * blockDim.x + threadIdx.x;
    int ktiles = K >> 5;
    int total  = (NPAD >> 4) * ktiles * 512;
    if (idx >= total) return;
    int j     = idx & 15;
    int lane  = (idx >> 4) & 31;
    int frag  = idx >> 9;
    int ntile = frag / ktiles;
    int ktile = frag % ktiles;
    int n = (ntile << 4) + (lane & 15);
    int k = (ktile << 5) + ((j >> 3) << 4) + ((lane >> 4) << 3) + (j & 7);
    Bsw[idx] = (n < N) ? (_Float16)W[(size_t)n * K + k] : (_Float16)0.f;
}

// ---------------------------------------------------------------------------
// WMMA GEMM: C[M x N] = A * W^T. A: f16 (MPAD x K) zero-padded; Bsw: f16
// pre-swizzled fragments. Each wave computes a 32x64 tile: 2 M-tiles x
// 4 N-tiles = 8 v_wmma per K-step with full A/B fragment reuse.
// ---------------------------------------------------------------------------
__device__ inline v16h cat16(v8h lo, v8h hi) {
    v16h r;
#pragma unroll
    for (int j = 0; j < 8; ++j) { r[j] = lo[j]; r[8 + j] = hi[j]; }
    return r;
}

__global__ void k_wmma_gemm(const _Float16* __restrict__ A,
                            const _Float16* __restrict__ Bsw,
                            float* __restrict__ C,
                            int Mvalid, int N, int NPAD, int K) {
    int wave = (blockIdx.x * blockDim.x + threadIdx.x) >> 5;
    int lane = threadIdx.x & 31;
    int tilesM2 = MPAD >> 5;          // 32-row strips
    int tilesN4 = NPAD >> 6;
    int tm  = wave / tilesN4;
    int tn4 = wave % tilesN4;
    if (tm >= tilesM2) return;        // wave-uniform
    int row    = tm << 5;
    int ktiles = K >> 5;

    const v8f vz = {0.f, 0.f, 0.f, 0.f, 0.f, 0.f, 0.f, 0.f};
    v8f acc[2][4] = {{vz, vz, vz, vz}, {vz, vz, vz, vz}};

    // A fragment base: lane layout (16-bit A 16x32): m = row + (lane&15),
    // elements 0..7 at k0 + kh, elements 8..15 at k0 + 16 + kh, kh = (lane>>4)*8
    int kh = (lane >> 4) * 8;
    const _Float16* A0 = A + (size_t)(row + (lane & 15)) * K + kh;
    const _Float16* A1 = A0 + (size_t)16 * K;
    const _Float16* Bp = Bsw + (size_t)(tn4 * 4) * ktiles * 512 + lane * 16;

    for (int kt = 0; kt < ktiles; ++kt) {
        int k0 = kt << 5;
        if (kt + 1 < ktiles) {
            __builtin_prefetch(A0 + k0 + 32, 0, 1);   // global_prefetch_b8
            __builtin_prefetch(A1 + k0 + 32, 0, 1);
        }
        v16h a0 = cat16(*(const v8h*)(A0 + k0), *(const v8h*)(A0 + k0 + 16));
        v16h a1 = cat16(*(const v8h*)(A1 + k0), *(const v8h*)(A1 + k0 + 16));
#pragma unroll
        for (int t = 0; t < 4; ++t) {
            const _Float16* bp = Bp + (size_t)(t * ktiles + kt) * 512;
            v16h bf = cat16(*(const v8h*)bp, *(const v8h*)(bp + 8));
            acc[0][t] = __builtin_amdgcn_wmma_f32_16x16x32_f16(
                false, a0, false, bf, (short)0, acc[0][t], false, false);
            acc[1][t] = __builtin_amdgcn_wmma_f32_16x16x32_f16(
                false, a1, false, bf, (short)0, acc[1][t], false, false);
        }
    }

    // C/D layout: lanes 0-15 -> M = r, lanes 16-31 -> M = 8 + r; N = lane&15
    int ml = (lane >> 4) * 8;
    int nl = lane & 15;
#pragma unroll
    for (int mt = 0; mt < 2; ++mt)
#pragma unroll
        for (int t = 0; t < 4; ++t) {
            int col = ((tn4 * 4 + t) << 4) + nl;
            if (col >= N) continue;
#pragma unroll
            for (int r = 0; r < 8; ++r) {
                int m = row + mt * 16 + ml + r;
                if (m < Mvalid) C[(size_t)m * N + col] = acc[mt][t][r];
            }
        }
}

// ---------------------------------------------------------------------------
// K3a: dt = softplus(zx[.,1280+h] + dt_bias); dA = exp(dt * -exp(A_log))
// ---------------------------------------------------------------------------
__global__ void k_dt(const float* __restrict__ zx,
                     const float* __restrict__ dt_bias,
                     const float* __restrict__ A_log,
                     float* __restrict__ dt, float* __restrict__ dA) {
    int idx = blockIdx.x * blockDim.x + threadIdx.x;
    if (idx >= MROWS * NHEADS) return;
    int hh = idx & 7;
    int bl = idx >> 3;
    float v  = zx[(size_t)bl * NPROJ + (DINNER + CONVDIM) + hh] + dt_bias[hh];
    float sp = (v > 20.f) ? v : log1pf(__expf(v));
    float a  = -__expf(A_log[hh]);
    dt[idx] = sp;
    dA[idx] = __expf(sp * a);
}

// ---------------------------------------------------------------------------
// K3b: depthwise causal conv (K=4, pad 3 left) over zx cols [512,1280) + SiLU
// ---------------------------------------------------------------------------
__global__ void k_dwconv_silu(const float* __restrict__ zx,
                              const float* __restrict__ w,
                              const float* __restrict__ bb,
                              float* __restrict__ xBCs) {
    int idx = blockIdx.x * blockDim.x + threadIdx.x;
    if (idx >= MROWS * CONVDIM) return;
    int ch = idx % CONVDIM;
    int bl = idx / CONVDIM;
    int l  = bl % LC;
    int b  = bl / LC;
    float s = bb[ch];
#pragma unroll
    for (int k = 0; k < 4; ++k) {
        int lk = l + k - 3;
        if (lk >= 0)
            s = fmaf(zx[((size_t)(b * LC + lk)) * NPROJ + DINNER + ch], w[ch * 4 + k], s);
    }
    float sig = 1.f / (1.f + __expf(-s));
    xBCs[(size_t)bl * CONVDIM + ch] = s * sig;
}

// ---------------------------------------------------------------------------
// K4: sequential SSM scan. 64 blocks = (b, head); 256 threads hold the
// 64x128 state in registers (32 floats/thread). B_t/C_t staged in LDS,
// dot over n reduced with wave32 shfl_xor.
// ---------------------------------------------------------------------------
__global__ void k_scan(const float* __restrict__ xBCs,
                       const float* __restrict__ dt,
                       const float* __restrict__ dA,
                       const float* __restrict__ D_skip,
                       float* __restrict__ y_ssm) {
    int b  = blockIdx.x >> 3;
    int hh = blockIdx.x & 7;
    int t  = threadIdx.x;
    int p  = t >> 2;
    int n0 = (t & 3) * 32;

    float hstate[32];
#pragma unroll
    for (int i = 0; i < 32; ++i) hstate[i] = 0.f;

    __shared__ float Bsh[DSTATE];
    __shared__ float Csh[DSTATE];
    float dsk = D_skip[hh];

    for (int l = 0; l < LC; ++l) {
        size_t rbl = (size_t)(b * LC + l);
        __syncthreads();                               // protect prev-iter reads
        if (t < 128) Bsh[t]       = xBCs[rbl * CONVDIM + DINNER + t];
        else         Csh[t - 128] = xBCs[rbl * CONVDIM + DINNER + DSTATE + (t - 128)];
        __syncthreads();

        float dA_t = dA[rbl * NHEADS + hh];
        float dt_t = dt[rbl * NHEADS + hh];
        float xp   = xBCs[rbl * CONVDIM + hh * HEADDIM + p];
        float coef = dt_t * xp;
        float accv = 0.f;
#pragma unroll
        for (int i = 0; i < 32; ++i) {
            hstate[i] = fmaf(hstate[i], dA_t, coef * Bsh[n0 + i]);
            accv      = fmaf(hstate[i], Csh[n0 + i], accv);
        }
        accv += __shfl_xor(accv, 1);
        accv += __shfl_xor(accv, 2);
        if ((t & 3) == 0)
            y_ssm[rbl * DINNER + hh * HEADDIM + p] = accv + dsk * xp;
    }
}

// ---------------------------------------------------------------------------
// K5: y = y_ssm * silu(z); RMSNorm over 512; * norm_w -> f16 GEMM2 A operand
// ---------------------------------------------------------------------------
__global__ void k_gate_rmsnorm(const float* __restrict__ zx,
                               const float* __restrict__ norm_w,
                               const float* __restrict__ y,
                               _Float16* __restrict__ y16) {
    int bl = blockIdx.x;
    int t  = threadIdx.x;
    __shared__ float red[8];
    __shared__ float rtot;

    float v[2];
    float ss = 0.f;
#pragma unroll
    for (int j = 0; j < 2; ++j) {
        int e = t * 2 + j;
        float z   = zx[(size_t)bl * NPROJ + e];
        float sig = 1.f / (1.f + __expf(-z));
        float val = y[(size_t)bl * DINNER + e] * z * sig;
        v[j] = val;
        ss  = fmaf(val, val, ss);
    }
#pragma unroll
    for (int m = 1; m < 32; m <<= 1) ss += __shfl_xor(ss, m);
    if ((t & 31) == 0) red[t >> 5] = ss;
    __syncthreads();
    if (t == 0) {
        float s = 0.f;
        for (int i = 0; i < 8; ++i) s += red[i];
        rtot = rsqrtf(s / (float)DINNER + 1e-5f);
    }
    __syncthreads();
    float r = rtot;
#pragma unroll
    for (int j = 0; j < 2; ++j) {
        int e = t * 2 + j;
        y16[(size_t)bl * DINNER + e] = (_Float16)(v[j] * r * norm_w[e]);
    }
}

// ---------------------------------------------------------------------------
// K7: conv1 (256->8, K=16, pad (7,8)) on y_out rows; out (b, 8, LC)
// ---------------------------------------------------------------------------
__global__ void k_conv_mid(const float* __restrict__ yin,
                           const float* __restrict__ w,
                           const float* __restrict__ bb,
                           float* __restrict__ y2) {
    int idx = blockIdx.x * blockDim.x + threadIdx.x;
    if (idx >= B_SZ * 8 * LC) return;
    int l = idx % LC;
    int r = idx / LC;
    int o = r & 7;
    int b = r >> 3;
    float s = bb[o];
    for (int k = 0; k < 16; ++k) {
        int lk = l + k - 7;
        if (lk < 0 || lk >= LC) continue;
        const float* yp = yin + (size_t)(b * LC + lk) * 256;
        const float* wp = w + (size_t)o * 256 * 16 + k;
        for (int c = 0; c < 256; ++c)
            s = fmaf(yp[c], wp[c * 16], s);
    }
    y2[idx] = s;
}

// ---------------------------------------------------------------------------
// K8: transposed conv (8->4, K=16, stride 8, pad 15): out (b, 4, 16384)
// ---------------------------------------------------------------------------
__global__ void k_convT(const float* __restrict__ y2,
                        const float* __restrict__ wT,
                        const float* __restrict__ bT,
                        float* __restrict__ out) {
    int idx = blockIdx.x * blockDim.x + threadIdx.x;
    if (idx >= B_SZ * 4 * L_IN) return;
    int tpos = idx & (L_IN - 1);
    int r    = idx >> 14;
    int i    = r & 3;
    int b    = r >> 2;
    float s = bT[i];
    int k0 = (15 - tpos) & 7;          // k s.t. (tpos + k - 15) % 8 == 0
#pragma unroll
    for (int kk = 0; kk < 2; ++kk) {
        int k = k0 + kk * 8;
        int j = tpos + k - 15;
        if (j < 0) continue;
        int src = j >> 3;
        if (src >= LC) continue;
#pragma unroll
        for (int o = 0; o < 8; ++o)
            s = fmaf(y2[((size_t)b * 8 + o) * LC + src], wT[((size_t)o * 4 + i) * 16 + (15 - k)], s);
    }
    out[idx] = s;
}

// ---------------------------------------------------------------------------
extern "C" void kernel_launch(void* const* d_in, const int* in_sizes, int n_in,
                              void* d_out, int out_size, void* d_ws, size_t ws_size,
                              hipStream_t stream) {
    (void)in_sizes; (void)n_in; (void)out_size; (void)ws_size;

    const float* x         = (const float*)d_in[0];
    const float* conv_w    = (const float*)d_in[1];
    const float* conv_b    = (const float*)d_in[2];
    const float* in_proj_w = (const float*)d_in[3];   // (1288, 256)
    const float* conv1d_w  = (const float*)d_in[4];   // (768, 1, 4)
    const float* conv1d_b  = (const float*)d_in[5];
    const float* dt_bias   = (const float*)d_in[6];
    const float* A_log     = (const float*)d_in[7];
    const float* D_skip    = (const float*)d_in[8];
    const float* norm_w    = (const float*)d_in[9];
    const float* out_proj_w= (const float*)d_in[10];  // (256, 512)
    const float* conv1_w   = (const float*)d_in[11];  // (8, 256, 16)
    const float* conv1_b   = (const float*)d_in[12];
    const float* convT_w   = (const float*)d_in[13];  // (8, 4, 16)
    const float* convT_b   = (const float*)d_in[14];
    float* out = (float*)d_out;

    // workspace layout (all sizes are multiples of 256 bytes)
    char* ws = (char*)d_ws;
    size_t off = 0;
    _Float16* h16   = (_Float16*)(ws + off); off += (size_t)MPAD * 256 * 2;
    float*    zx    = (float*)   (ws + off); off += (size_t)MROWS * NPROJ * 4;
    _Float16* Bsw1  = (_Float16*)(ws + off); off += (size_t)NPROJ_PAD * 256 * 2;
    _Float16* Bsw2  = (_Float16*)(ws + off); off += (size_t)256 * 512 * 2;
    float*    xBCs  = (float*)   (ws + off); off += (size_t)MROWS * CONVDIM * 4;
    float*    dtb   = (float*)   (ws + off); off += (size_t)MROWS * NHEADS * 4;
    float*    dAb   = (float*)   (ws + off); off += (size_t)MROWS * NHEADS * 4;
    float*    y_ssm = (float*)   (ws + off); off += (size_t)MROWS * DINNER * 4;
    _Float16* y16   = (_Float16*)(ws + off); off += (size_t)MPAD * DINNER * 2;
    float*    y2    = (float*)   (ws + off); off += (size_t)B_SZ * 8 * LC * 4;
    float*    y_out = zx;   // alias: zx dead after gate/rmsnorm, before GEMM2

    // K1: front conv + relu (f16 A operand) + zero pad rows
    {
        int n = MROWS * 256;
        k_conv_front<<<(n + 255) / 256, 256, 0, stream>>>(x, conv_w, conv_b, h16);
        int pad = (MPAD - MROWS) * 256;
        k_zero_f16<<<(pad + 255) / 256, 256, 0, stream>>>(h16 + (size_t)MROWS * 256, pad);
        int pad2 = (MPAD - MROWS) * DINNER;
        k_zero_f16<<<(pad2 + 255) / 256, 256, 0, stream>>>(y16 + (size_t)MROWS * DINNER, pad2);
    }
    // B-operand pre-swizzle for both GEMMs
    {
        int n1 = (NPROJ_PAD >> 4) * (256 >> 5) * 512;
        k_swizzleB<<<(n1 + 255) / 256, 256, 0, stream>>>(in_proj_w, Bsw1, NPROJ, 256, NPROJ_PAD);
        int n2 = (256 >> 4) * (512 >> 5) * 512;
        k_swizzleB<<<(n2 + 255) / 256, 256, 0, stream>>>(out_proj_w, Bsw2, 256, 512, 256);
    }
    // K2: in_proj GEMM (M=16376, N=1288, K=256), 8 v_wmma / K-step / wave
    {
        int waves = (MPAD >> 5) * (NPROJ_PAD >> 6);
        int threads = waves * 32;
        k_wmma_gemm<<<(threads + 255) / 256, 256, 0, stream>>>(h16, Bsw1, zx, MROWS, NPROJ, NPROJ_PAD, 256);
    }
    // K3: dt/dA + depthwise conv + silu
    {
        int n = MROWS * NHEADS;
        k_dt<<<(n + 255) / 256, 256, 0, stream>>>(zx, dt_bias, A_log, dtb, dAb);
        int m = MROWS * CONVDIM;
        k_dwconv_silu<<<(m + 255) / 256, 256, 0, stream>>>(zx, conv1d_w, conv1d_b, xBCs);
    }
    // K4: sequential SSM scan, 64 blocks = (b, head)
    k_scan<<<B_SZ * NHEADS, 256, 0, stream>>>(xBCs, dtb, dAb, D_skip, y_ssm);
    // K5: gate + rmsnorm -> f16 A operand for GEMM2
    k_gate_rmsnorm<<<MROWS, 256, 0, stream>>>(zx, norm_w, y_ssm, y16);
    // K6: out_proj GEMM (M=16376, N=256, K=512)
    {
        int waves = (MPAD >> 5) * (256 >> 6);
        int threads = waves * 32;
        k_wmma_gemm<<<(threads + 255) / 256, 256, 0, stream>>>(y16, Bsw2, y_out, MROWS, 256, 256, 512);
    }
    // K7: conv1 (256 -> 8)
    {
        int n = B_SZ * 8 * LC;
        k_conv_mid<<<(n + 255) / 256, 256, 0, stream>>>(y_out, conv1_w, conv1_b, y2);
    }
    // K8: transposed conv (8 -> 4, stride 8)
    {
        int n = B_SZ * 4 * L_IN;
        k_convT<<<(n + 255) / 256, 256, 0, stream>>>(y2, convT_w, convT_b, out);
    }
}
